// VectorQuantizer_75230647157531
// MI455X (gfx1250) — matrix operand order repaired and checked by reading, not compiled, not run on hardware
//
#include <hip/hip_runtime.h>
#include <math.h>

typedef __attribute__((ext_vector_type(16))) __bf16 v16bf;
typedef __attribute__((ext_vector_type(8)))  float  v8f;
typedef __attribute__((ext_vector_type(2)))  float  v2f;
typedef __attribute__((ext_vector_type(4)))  unsigned u32x4;
typedef __attribute__((ext_vector_type(8)))  int      i32x8;
typedef __attribute__((ext_vector_type(4)))  int      i32x4;

#define BATCH 16
#define C_IN  256
#define HW    4096
#define DIM   128
#define KCB   1024
#define NPOS  65536      // BATCH * HW
#define ROWS  128        // rows (positions) per workgroup
#define BETA_ 0.25f
#define EPS_  1e-8f
#define CBPITCH 129      // padded LDS pitch (floats) produced by TDM padding

// counter-based uniform in (0,1)  (JAX threefry not reproducible; stochastic path)
__device__ __forceinline__ float urand01(unsigned x) {
    x ^= x >> 17; x *= 0xed5ad4bbu; x ^= x >> 11; x *= 0xac4c1b51u;
    x ^= x >> 15; x *= 0x31848babu; x ^= x >> 14;
    return ((float)(x >> 8) + 0.5f) * (1.0f / 16777216.0f);
}

// ---------------------------------------------------------------------------
// Kernel 0: per-code squared norms of the codebook
// ---------------------------------------------------------------------------
__global__ __launch_bounds__(256) void cnorm_kernel(const float* __restrict__ cb,
                                                    float* __restrict__ cnorm) {
    int k = blockIdx.x * 256 + threadIdx.x;
    if (k < KCB) {
        const float* r = cb + (size_t)k * DIM;
        float s = 0.f;
        #pragma unroll
        for (int d = 0; d < DIM; d += 4) {
            float4 v = *(const float4*)(r + d);
            s += v.x * v.x + v.y * v.y + v.z * v.z + v.w * v.w;
        }
        cnorm[k] = s;
    }
}

// ---------------------------------------------------------------------------
// Kernel 1: fused dual 1x1-conv GEMM (bf16 WMMA) + stick-breaking mask + KLD
//   grid 512 x 256 threads (8 waves). Each block: 128 positions x 128 dims.
//   LDS (dynamic, 66560B): [A 128x36 bf16 | Benc 128x36 | Bpvnd 128x36]
//                          aliased with swizzled f32 stage [128][128]
// ---------------------------------------------------------------------------
__global__ __launch_bounds__(256) void enc_mask_kernel(
    const float* __restrict__ emb,   const float* __restrict__ w_enc,
    const float* __restrict__ b_enc, const float* __restrict__ w_pvnd,
    const float* __restrict__ b_pvnd,const float* __restrict__ pv,
    float* __restrict__ latents,     float* __restrict__ kld_accum) {
    extern __shared__ char smem[];
    __bf16* sA    = (__bf16*)smem;               // [128][36] bf16
    __bf16* sBe   = (__bf16*)(smem + 9216);      // [128][36] bf16
    __bf16* sBp   = (__bf16*)(smem + 18432);     // [128][36] bf16
    float*  stage = (float*)smem;                // [128*128] f32 (reuse after GEMM)
    int*    sIstar= (int*)  (smem + 65536);      // [128]
    float*  sKld  = (float*)(smem + 66048);      // [128]

    const int tid  = threadIdx.x;
    const int wv   = tid >> 5;
    const int lane = tid & 31;
    const int lo   = lane & 15;
    const int hi   = lane >> 4;
    const int n0   = blockIdx.x * ROWS;
    const int b    = n0 >> 12;
    const int hw0  = n0 & (HW - 1);

    v8f accF[8], accP[8];
    #pragma unroll
    for (int t = 0; t < 8; ++t) {
        accF[t] = (v8f){0.f,0.f,0.f,0.f,0.f,0.f,0.f,0.f};
        accP[t] = (v8f){0.f,0.f,0.f,0.f,0.f,0.f,0.f,0.f};
    }

    for (int kc = 0; kc < 8; ++kc) {             // K = 256 in chunks of 32
        __syncthreads();
        { // stage A tile: 128 rows x 32 c  (coalesced 128-float runs per c)
            int c = tid >> 3, sub = tid & 7;
            const float* src = emb + ((size_t)(b * C_IN + kc * 32 + c) * HW) + hw0 + sub * 16;
            #pragma unroll
            for (int i = 0; i < 16; i += 4) {
                float4 v = *(const float4*)(src + i);
                int r = sub * 16 + i;
                sA[(r + 0) * 36 + c] = (__bf16)v.x;
                sA[(r + 1) * 36 + c] = (__bf16)v.y;
                sA[(r + 2) * 36 + c] = (__bf16)v.z;
                sA[(r + 3) * 36 + c] = (__bf16)v.w;
            }
        }
        { // stage both weight tiles: 128 d x 32 c
            int d = tid >> 1, half = tid & 1;
            const float* se = w_enc  + (size_t)d * C_IN + kc * 32 + half * 16;
            const float* sp = w_pvnd + (size_t)d * C_IN + kc * 32 + half * 16;
            #pragma unroll
            for (int i = 0; i < 16; i += 4) {
                float4 v = *(const float4*)(se + i);
                float4 u = *(const float4*)(sp + i);
                int c = half * 16 + i;
                sBe[d * 36 + c + 0] = (__bf16)v.x;  sBe[d * 36 + c + 1] = (__bf16)v.y;
                sBe[d * 36 + c + 2] = (__bf16)v.z;  sBe[d * 36 + c + 3] = (__bf16)v.w;
                sBp[d * 36 + c + 0] = (__bf16)u.x;  sBp[d * 36 + c + 1] = (__bf16)u.y;
                sBp[d * 36 + c + 2] = (__bf16)u.z;  sBp[d * 36 + c + 3] = (__bf16)u.w;
            }
        }
        __syncthreads();

        // A fragment: row = 16*wv + lo ; K(j) per ISA 16-bit A 16x32 layout
        union { v16bf v; unsigned u[8]; } a;
        const int arow = 16 * wv + lo;
        #pragma unroll
        for (int j = 0; j < 8; ++j) {
            int k = (j < 4 ? 2 * j : 16 + 2 * (j - 4)) + 8 * hi;
            a.u[j] = *(const unsigned*)((const char*)sA + ((size_t)arow * 36 + k) * 2);
        }
        #pragma unroll
        for (int ct = 0; ct < 8; ++ct) {
            union { v16bf v; unsigned u[8]; } be, bp;
            const int dn = ct * 16 + lo;
            #pragma unroll
            for (int j = 0; j < 8; ++j) {
                int k = (j < 4 ? 2 * j : 16 + 2 * (j - 4)) + 8 * hi;   // 16-bit B 32x16 layout
                be.u[j] = *(const unsigned*)((const char*)sBe + ((size_t)dn * 36 + k) * 2);
                bp.u[j] = *(const unsigned*)((const char*)sBp + ((size_t)dn * 36 + k) * 2);
            }
            accF[ct] = __builtin_amdgcn_wmma_f32_16x16x32_bf16(
                false, a.v, false, be.v, (short)0, accF[ct], false, false);
            accP[ct] = __builtin_amdgcn_wmma_f32_16x16x32_bf16(
                false, a.v, false, bp.v, (short)0, accP[ct], false, false);
        }
    }
    __syncthreads();

    // bias add; dump p_vnd to swizzled stage [col][ (row+col)&127 ]
    #pragma unroll
    for (int ct = 0; ct < 8; ++ct) {
        int col = ct * 16 + lo;
        float bbe = b_enc[col], bbp = b_pvnd[col];
        #pragma unroll
        for (int j = 0; j < 8; ++j) {
            int r = 16 * wv + j + 8 * hi;          // C/D layout: M = j + 8*(lane>>4)
            accF[ct][j] += bbe;
            stage[col * 128 + ((r + col) & 127)] = accP[ct][j] + bbp;
        }
    }
    __syncthreads();

    // per-row sequential stick-breaking + gumbel argmax + KLD
    if (tid < 128) {
        const int r = tid;
        float cum = 1.f, best = -3.4e38f, kld = 0.f;
        int   besti = 0;
        for (int d = 0; d < DIM; ++d) {
            float qv;
            if (d < DIM - 1) {
                float p = stage[(d + 1) * 128 + ((r + d + 1) & 127)];
                p = fminf(5.f, fmaxf(-5.f, p));
                float bta = 1.f / (1.f + __expf(-p));
                qv  = cum * (1.f - bta);
                cum *= bta;
            } else {
                qv = cum;
            }
            kld += qv * __logf(qv / pv[d] + EPS_);
            unsigned seed = (unsigned)((n0 + r) * DIM + d) * 2654435761u + 0x9e3779b9u;
            float g = -__logf(-__logf(urand01(seed)));
            float val = qv + g;                    // TAU = 1
            if (val > best) { best = val; besti = d; }
        }
        sIstar[r] = besti;
        sKld[r]   = kld;
    }
    __syncthreads();
    if (tid == 0) {
        float s = 0.f;
        for (int i = 0; i < 128; ++i) s += sKld[i];
        atomicAdd(kld_accum, s);
    }

    // masked latents -> workspace  (s_vnd: ones through argmax index)
    #pragma unroll
    for (int ct = 0; ct < 8; ++ct) {
        int col = ct * 16 + lo;
        #pragma unroll
        for (int j = 0; j < 8; ++j) {
            int r  = 16 * wv + j + 8 * hi;
            int is = sIstar[r];
            latents[(size_t)(n0 + r) * DIM + col] = (col <= is) ? accF[ct][j] : 0.f;
        }
    }
}

// ---------------------------------------------------------------------------
// Kernel 2: VQ distances (f32 WMMA 16x16x4) + argmin + gather + NCHW store
//   Codebook chunks are DMA'd into LDS by the Tensor Data Mover with
//   1-DWORD padding every 128 DWORDs => conflict-free pitch-129 layout.
//   grid 512 x 256. LDS 67072B: 66048B padded chunk buffer (codebook ->
//   candidates -> quant stage) + xnorm[128] + istar[128]
// ---------------------------------------------------------------------------
__global__ __launch_bounds__(256) void vq_kernel(
    const float* __restrict__ latents, const float* __restrict__ codebook,
    const float* __restrict__ cnorm,   float* __restrict__ out,
    float* __restrict__ mse_accum) {
    extern __shared__ char smem[];
    float* big = (float*)smem;               // 16512 floats (pitch-129 chunk image)
    float* sXn = (float*)(smem + 66048);     // [128]
    int*   sIs = (int*)  (smem + 66560);     // [128]

    const int tid  = threadIdx.x;
    const int wv   = tid >> 5;
    const int lane = tid & 31;
    const int lo   = lane & 15;
    const int hi   = lane >> 4;
    const int n0   = blockIdx.x * ROWS;
    const int brow = 16 * wv + lo;           // A row: f32 A 16x4 layout, M = lane&15

    // Load all A fragments once (kept in regs for all 8 code chunks) + x-norm
    v2f ar[32];
    float xn = 0.f;
    {
        const float* lp = latents + (size_t)(n0 + brow) * DIM + 2 * hi;
        #pragma unroll
        for (int s = 0; s < 32; ++s) {       // K(j) = 4s + 2*(lane>>4) + j
            float2 v = *(const float2*)(lp + 4 * s);
            ar[s].x = v.x; ar[s].y = v.y;
            xn += v.x * v.x + v.y * v.y;
        }
    }
    xn += __shfl_xor(xn, 16, 32);
    if (hi == 0) sXn[brow] = xn;

    // TDM descriptor pieces invariant across chunks
    const unsigned lds_off = (unsigned)(uintptr_t)(void*)big;   // LDS byte offset

    float minv[8]; int mini[8];
    #pragma unroll
    for (int j = 0; j < 8; ++j) { minv[j] = 3.4e38f; mini[j] = 0; }

    for (int ch = 0; ch < 8; ++ch) {         // 1024 codes in chunks of 128
        const int cbase = ch * 128;
        __syncthreads();                     // previous chunk fully consumed
        if (wv == 0) {
            // Tensor Data Mover: 2D tile [128 rows x 128 f32], pad 1 DWORD per
            // 128 DWORDs -> LDS pitch 129 floats (bank-conflict-free frags).
            unsigned long long gaddr =
                (unsigned long long)(uintptr_t)(codebook + (size_t)cbase * DIM);
            u32x4 g0;
            g0[0] = 1u;                                        // count=1 (valid)
            g0[1] = lds_off;                                   // lds_addr
            g0[2] = (unsigned)(gaddr & 0xffffffffu);           // global_addr lo
            g0[3] = (unsigned)((gaddr >> 32) & 0x01ffffffu)    // global_addr hi
                    | (2u << 30);                              // type = 2 (image)
            i32x8 g1;
            g1[0] = (2 << 16)      // data_size = 4B
                  | (1 << 20)      // pad_enable
                  | (6 << 22);     // pad_interval = 128 DWORDs (pad_amount=1 DW)
            g1[1] = (DIM & 0xffff) << 16;      // tensor_dim0 = 128  (lo16)
            g1[2] = (128 & 0xffff) << 16;      // dim0 hi=0 | tensor_dim1 lo16 = 128
            g1[3] = (128 & 0xffff) << 16;      // dim1 hi=0 | tile_dim0 = 128
            g1[4] = 128;                       // tile_dim1 = 128, tile_dim2 = 0
            g1[5] = DIM;                       // tensor_dim0_stride = 128 (lo32)
            g1[6] = 0;
            g1[7] = 0;
            i32x4 gz4 = (i32x4){0, 0, 0, 0};   // groups 2/3 unused (2D tensor)
            i32x8 gz8 = (i32x8){0, 0, 0, 0, 0, 0, 0, 0};
            __builtin_amdgcn_tensor_load_to_lds(g0, g1, gz4, gz4, gz8, 0);
            __builtin_amdgcn_s_wait_tensorcnt(0);
        }
        __syncthreads();                     // chunk visible to all waves

        #pragma unroll
        for (int ct = 0; ct < 8; ++ct) {
            v8f acc = (v8f){0.f,0.f,0.f,0.f,0.f,0.f,0.f,0.f};
            const int code = ct * 16 + lo;   // f32 B 4x16 layout: N = lane&15
            #pragma unroll
            for (int s = 0; s < 32; ++s) {
                int k = 4 * s + 2 * hi;
                v2f bf;
                bf.x = big[code * CBPITCH + k + 0];
                bf.y = big[code * CBPITCH + k + 1];
                acc = __builtin_amdgcn_wmma_f32_16x16x4_f32(
                    false, ar[s], false, bf, (short)0, acc, false, false);
            }
            const int gcode = cbase + code;
            const float cn  = cnorm[gcode];
            #pragma unroll
            for (int j = 0; j < 8; ++j) {
                float score = cn - 2.f * acc[j];   // dist - ||x||^2
                if (score < minv[j]) { minv[j] = score; mini[j] = gcode; }
            }
        }
    }
    __syncthreads();

    // per-row candidate reduction (16 lane-classes per row)
    float* cv = big;                 // [128][16]
    int*   ci = (int*)(big + 2048);  // [128][16]
    #pragma unroll
    for (int j = 0; j < 8; ++j) {
        int r = 16 * wv + j + 8 * hi;
        cv[r * 16 + lo] = minv[j];
        ci[r * 16 + lo] = mini[j];
    }
    __syncthreads();
    if (tid < 128) {
        float bv = 3.4e38f; int bi = 0x7fffffff;
        for (int i = 0; i < 16; ++i) {
            float v = cv[tid * 16 + i]; int ix = ci[tid * 16 + i];
            if (v < bv || (v == bv && ix < bi)) { bv = v; bi = ix; }
        }
        sIs[tid] = bi;
        atomicAdd(mse_accum, bv + sXn[tid]);     // = ||quant - latent||^2 row-sum
    }
    __syncthreads();

    // gather quant rows into swizzled stage [d][ (row+d)&127 ]
    {
        int row = tid >> 1, dbase = (tid & 1) * 64;
        const float* src = codebook + (size_t)sIs[row] * DIM + dbase;
        #pragma unroll
        for (int i = 0; i < 64; i += 4) {
            float4 v = *(const float4*)(src + i);
            int d = dbase + i;
            big[(d + 0) * 128 + ((row + d + 0) & 127)] = v.x;
            big[(d + 1) * 128 + ((row + d + 1) & 127)] = v.y;
            big[(d + 2) * 128 + ((row + d + 2) & 127)] = v.z;
            big[(d + 3) * 128 + ((row + d + 3) & 127)] = v.w;
        }
    }
    __syncthreads();

    // transposed, fully-coalesced NCHW store
    const int b = n0 >> 12, hw0 = n0 & (HW - 1);
    #pragma unroll 4
    for (int it = 0; it < 64; ++it) {
        int idx = it * 256 + tid;
        int d = idx >> 7, t2 = idx & 127;
        out[((size_t)(b * DIM + d)) * HW + hw0 + t2] = big[d * 128 + ((t2 + d) & 127)];
    }
}

// ---------------------------------------------------------------------------
// Kernel 3: scalar loss  vq_loss = mse*(1+beta) + kld*beta
// ---------------------------------------------------------------------------
__global__ void finalize_kernel(const float* __restrict__ acc, float* __restrict__ out) {
    float kld = acc[0] / (float)NPOS;
    float mse = acc[1] / ((float)NPOS * (float)DIM);
    out[(size_t)NPOS * DIM] = mse * (1.f + BETA_) + kld * BETA_;
}

// ---------------------------------------------------------------------------
extern "C" void kernel_launch(void* const* d_in, const int* in_sizes, int n_in,
                              void* d_out, int out_size, void* d_ws, size_t ws_size,
                              hipStream_t stream) {
    const float* emb      = (const float*)d_in[0];
    const float* w_enc    = (const float*)d_in[1];
    const float* b_enc    = (const float*)d_in[2];
    const float* w_pvnd   = (const float*)d_in[3];
    const float* b_pvnd   = (const float*)d_in[4];
    const float* codebook = (const float*)d_in[5];
    const float* pv       = (const float*)d_in[6];
    float* out = (float*)d_out;

    float* wsf     = (float*)d_ws;
    float* latents = wsf + 64;                       // N*D floats (32 MiB)
    float* cnorm   = latents + (size_t)NPOS * DIM;   // K floats

    (void)hipMemsetAsync(d_ws, 0, 256, stream);      // zero loss accumulators

    cnorm_kernel<<<dim3(KCB / 256), dim3(256), 0, stream>>>(codebook, cnorm);
    enc_mask_kernel<<<dim3(NPOS / ROWS), dim3(256), 66560, stream>>>(
        emb, w_enc, b_enc, w_pvnd, b_pvnd, pv, latents, wsf + 0);
    vq_kernel<<<dim3(NPOS / ROWS), dim3(256), 67072, stream>>>(
        latents, codebook, cnorm, out, wsf + 1);
    finalize_kernel<<<1, 1, 0, stream>>>(wsf, out);
}
